// ELKBlock_25417616458392
// MI455X (gfx1250) — compile-verified, matrix operand order checked by reading, not run on hardware
//
#include <hip/hip_runtime.h>
#include <hip/hip_bf16.h>

#define INC 128
#define K27 27

typedef __attribute__((ext_vector_type(16))) _Float16 v16h;
typedef __attribute__((ext_vector_type(8)))  _Float16 v8h;
typedef __attribute__((ext_vector_type(8)))  float    v8f;

// Load a 16-half A/B operand fragment for v_wmma_f32_16x16x32_f16.
// CDNA5 16-bit 16x32 layout: lanes 0-15 hold K=[0..7] (VGPR0-3) and K=[16..23]
// (VGPR4-7); lanes 16-31 hold K=[8..15] and K=[24..31]. `base` already includes
// the per-lane +8 for lanes 16-31.
__device__ __forceinline__ v16h load_frag16(const _Float16* row, int base) {
    v8h lo = *(const v8h*)(row + base);
    v8h hi = *(const v8h*)(row + base + 16);
    return __builtin_shufflevector(lo, hi, 0,1,2,3,4,5,6,7,8,9,10,11,12,13,14,15);
}

__device__ __forceinline__ void atomic_fadd(float* p, float v) {
    __hip_atomic_fetch_add(p, v, __ATOMIC_RELAXED, __HIP_MEMORY_SCOPE_AGENT);
}

// ---------------------------------------------------------------------------
// K0: convert F -> f16 (pad row N = 0), w_pre -> f16 (layout [d][c]),
//     w_conv -> f16 transposed to [tap][d][c]
// ---------------------------------------------------------------------------
__global__ void k_convert(const float* __restrict__ F, const float* __restrict__ w_pre,
                          const float* __restrict__ w_conv,
                          _Float16* __restrict__ F16, _Float16* __restrict__ Wpre16,
                          _Float16* __restrict__ WconvT16, int N) {
    const int total_f = (N + 1) * INC;
    const int total   = total_f + INC * INC + K27 * INC * INC;
    for (int i = blockIdx.x * blockDim.x + threadIdx.x; i < total;
         i += gridDim.x * blockDim.x) {
        if (i < total_f) {
            F16[i] = (_Float16)(i < N * INC ? F[i] : 0.0f);
        } else if (i < total_f + INC * INC) {
            int j = i - total_f;
            Wpre16[j] = (_Float16)w_pre[j];        // already [d][c]
        } else {
            int j = i - total_f - INC * INC;
            int k = j >> 14;                        // /(128*128)
            int rem = j & 16383;
            int d = rem >> 7, c = rem & 127;
            WconvT16[j] = (_Float16)w_conv[(size_t)k * (INC * INC) + c * INC + d];
        }
    }
}

// ---------------------------------------------------------------------------
// K1: zero the aux segment-sum buffer (num_aux read on device; graph-safe)
// ---------------------------------------------------------------------------
__global__ void k_zero_aux(float* __restrict__ aux, const int* __restrict__ num_aux_p) {
    int total = num_aux_p[0] * (3 * INC);
    int i = blockIdx.x * blockDim.x + threadIdx.x;
    if (i < total) aux[i] = 0.0f;
}

// ---------------------------------------------------------------------------
// K2: pre_mix GEMM (F @ w_pre^T) + LayerNorm -> F_input (stored in d_out)
//     One 16-row tile per block; wave w owns output columns [16w, 16w+16).
// ---------------------------------------------------------------------------
__global__ __launch_bounds__(256) void k_premix(const _Float16* __restrict__ F16,
                                                const _Float16* __restrict__ Wpre16,
                                                const float* __restrict__ g_pre,
                                                const float* __restrict__ b_pre,
                                                float* __restrict__ F_input, int N) {
    __shared__ float sF[16][INC];
    __shared__ float s1[16][16], s2[16][16];
    __shared__ float sm[16], si[16];

    const int tid  = threadIdx.x;
    const int lane = tid & 31;
    const int wave = tid >> 5;
    const int n0   = blockIdx.x * 16;
    const int mrow = lane & 15;
    const int khi  = (lane >> 4) << 3;   // +8 for upper half-wave

    const _Float16* arow = F16 + (size_t)(n0 + mrow) * INC;
    const _Float16* brow = Wpre16 + (size_t)(wave * 16 + mrow) * INC;

    v8f acc = {};
#pragma unroll
    for (int ks = 0; ks < 4; ++ks) {
        int k0 = ks * 32 + khi;
        v16h a = load_frag16(arow, k0);
        v16h b = load_frag16(brow, k0);
        acc = __builtin_amdgcn_wmma_f32_16x16x32_f16(false, a, false, b,
                                                     (short)0, acc, false, false);
    }
    // D layout: VGPR r -> (M = r + 8*(lane>=16), Ncol = lane&15)
    const int mbase = (lane >> 4) << 3;
    const int ch    = wave * 16 + (lane & 15);
#pragma unroll
    for (int r = 0; r < 8; ++r) sF[mbase + r][ch] = acc[r];
    __syncthreads();

    {
        int row = tid & 15, g = tid >> 4;
        float a1 = 0.f, a2 = 0.f;
#pragma unroll
        for (int j = 0; j < 8; ++j) { float x = sF[row][g * 8 + j]; a1 += x; a2 += x * x; }
        s1[row][g] = a1; s2[row][g] = a2;
    }
    __syncthreads();
    if (tid < 16) {
        float su = 0.f, sq = 0.f;
#pragma unroll
        for (int g = 0; g < 16; ++g) { su += s1[tid][g]; sq += s2[tid][g]; }
        float m = su * (1.0f / INC);
        float v = sq * (1.0f / INC) - m * m;
        sm[tid] = m; si[tid] = rsqrtf(v + 1e-6f);
    }
    __syncthreads();
#pragma unroll
    for (int j = 0; j < 8; ++j) {
        int e = tid + j * 256;
        int row = e >> 7, c = e & 127;
        float y = (sF[row][c] - sm[row]) * si[row] * g_pre[c] + b_pre[c];
        F_input[(size_t)(n0 + row) * INC + c] = y;
    }
}

// ---------------------------------------------------------------------------
// K3: scatter cat = [F_in*cos, F_in*sin, F_in*pos] into aux via fadd atomics
// ---------------------------------------------------------------------------
__global__ void k_scatter(const int* __restrict__ coords, const int* __restrict__ aux_idx,
                          const float* __restrict__ alpha, const float* __restrict__ w_pos,
                          const float* __restrict__ F_input, float* __restrict__ aux, int N) {
    int gid = blockIdx.x * blockDim.x + threadIdx.x;
    if (gid >= N * INC) return;
    int n = gid >> 7, c = gid & 127;
    float cx = (float)coords[n * 3 + 0];
    float cy = (float)coords[n * 3 + 1];
    float cz = (float)coords[n * 3 + 2];
    float pos = (cx * w_pos[c * 3 + 0] + cy * w_pos[c * 3 + 1] + cz * w_pos[c * 3 + 2]) * alpha[c];
    float fi = F_input[gid];
    float sn = sinf(pos), cs = cosf(pos);
    float* p = aux + (size_t)aux_idx[n] * (3 * INC) + c;
    atomic_fadd(p,            fi * cs);
    atomic_fadd(p + INC,      fi * sn);
    atomic_fadd(p + 2 * INC,  fi * pos);
}

// ---------------------------------------------------------------------------
// K4: fused 27-tap gathered WMMA conv (32 rows/block, M-blocked: each wave
//     computes a 32x16 output tile sharing one B fragment per 2 WMMAs)
//     + aux broadcast + dual LayerNorm + ReLU
// ---------------------------------------------------------------------------
__global__ __launch_bounds__(256) void k_final(const _Float16* __restrict__ F16,
                                               const _Float16* __restrict__ WconvT16,
                                               const int* __restrict__ nbr_idx,
                                               const int* __restrict__ coords,
                                               const int* __restrict__ aux_idx,
                                               const float* __restrict__ counts_v,
                                               const float* __restrict__ alpha,
                                               const float* __restrict__ w_pos,
                                               const float* __restrict__ F_input,
                                               const float* __restrict__ aux,
                                               const float* __restrict__ g_local,
                                               const float* __restrict__ b_local,
                                               const float* __restrict__ g_norm,
                                               const float* __restrict__ b_norm,
                                               float* __restrict__ out, int N) {
    __shared__ float sLoc[32][INC];
    __shared__ float sNew[32][INC];
    __shared__ float s1[2][32][8], s2[2][32][8];
    __shared__ float sm[2][32], si[2][32];

    const int tid  = threadIdx.x;
    const int lane = tid & 31;
    const int wave = tid >> 5;
    const int n0   = blockIdx.x * 32;          // 32 rows per block (100000 % 32 == 0)
    const int mrow = lane & 15;
    const int khi  = (lane >> 4) << 3;
    const int drow = wave * 16 + mrow;         // weight row d for this lane's column

    // --- sparse 3x3x3 conv: two M-tiles per wave, shared B fragment ---
    v8f accA = {}, accB = {};
    for (int tap = 0; tap < K27; ++tap) {
        int ia = nbr_idx[(size_t)(n0 + mrow) * K27 + tap];        // pad idx N -> zero row
        int ib = nbr_idx[(size_t)(n0 + 16 + mrow) * K27 + tap];
        const _Float16* arA = F16 + (size_t)ia * INC;
        const _Float16* arB = F16 + (size_t)ib * INC;
        const _Float16* br  = WconvT16 + ((size_t)tap * INC + drow) * INC;
#pragma unroll
        for (int ks = 0; ks < 4; ++ks) {
            int k0 = ks * 32 + khi;
            v16h b  = load_frag16(br,  k0);
            v16h aA = load_frag16(arA, k0);
            v16h aB = load_frag16(arB, k0);
            accA = __builtin_amdgcn_wmma_f32_16x16x32_f16(false, aA, false, b,
                                                          (short)0, accA, false, false);
            accB = __builtin_amdgcn_wmma_f32_16x16x32_f16(false, aB, false, b,
                                                          (short)0, accB, false, false);
        }
    }
    {
        const int mbase = (lane >> 4) << 3;
        const int ch    = wave * 16 + (lane & 15);
#pragma unroll
        for (int r = 0; r < 8; ++r) {
            sLoc[mbase + r][ch]      = accA[r];
            sLoc[16 + mbase + r][ch] = accB[r];
        }
    }

    // --- aux broadcast back: new_raw = v0*cos + v1*sin + (v2 - F_in*pos) ---
#pragma unroll
    for (int j = 0; j < 16; ++j) {
        int e = tid + j * 256;                 // 0..4095
        int row = e >> 7, c = e & 127;
        int n = n0 + row;
        float cx = (float)coords[n * 3 + 0];
        float cy = (float)coords[n * 3 + 1];
        float cz = (float)coords[n * 3 + 2];
        float pos = (cx * w_pos[c * 3 + 0] + cy * w_pos[c * 3 + 1] + cz * w_pos[c * 3 + 2]) * alpha[c];
        float sn = sinf(pos), cs = cosf(pos);
        float ic = 1.0f / counts_v[n];
        const float* p = aux + (size_t)aux_idx[n] * (3 * INC) + c;
        float v0 = p[0] * ic, v1 = p[INC] * ic, v2 = p[2 * INC] * ic;
        float fi = F_input[(size_t)n * INC + c];
        sNew[row][c] = v0 * cs + v1 * sn + (v2 - fi * pos);
    }
    __syncthreads();

    // --- dual LayerNorm reductions (32 rows x 8 groups of 16 channels) ---
    {
        int row = tid >> 3, g = tid & 7;
        float a1 = 0.f, a2 = 0.f, c1 = 0.f, c2 = 0.f;
#pragma unroll
        for (int j = 0; j < 16; ++j) {
            float x = sLoc[row][g * 16 + j]; a1 += x; a2 += x * x;
            float y = sNew[row][g * 16 + j]; c1 += y; c2 += y * y;
        }
        s1[0][row][g] = a1; s2[0][row][g] = a2;
        s1[1][row][g] = c1; s2[1][row][g] = c2;
    }
    __syncthreads();
    if (tid < 64) {
        int which = tid >> 5, row = tid & 31;
        float su = 0.f, sq = 0.f;
#pragma unroll
        for (int g = 0; g < 8; ++g) { su += s1[which][row][g]; sq += s2[which][row][g]; }
        float m = su * (1.0f / INC);
        float v = sq * (1.0f / INC) - m * m;
        sm[which][row] = m; si[which][row] = rsqrtf(v + 1e-6f);
    }
    __syncthreads();

    // --- combine: relu(LN(new) + LN(local)) ---
#pragma unroll
    for (int j = 0; j < 16; ++j) {
        int e = tid + j * 256;
        int row = e >> 7, c = e & 127;
        float ln_l = (sLoc[row][c] - sm[0][row]) * si[0][row] * g_local[c] + b_local[c];
        float ln_n = (sNew[row][c] - sm[1][row]) * si[1][row] * g_norm[c] + b_norm[c];
        float r = ln_n + ln_l;
        out[(size_t)(n0 + row) * INC + c] = r > 0.0f ? r : 0.0f;
    }
}

// ---------------------------------------------------------------------------
extern "C" void kernel_launch(void* const* d_in, const int* in_sizes, int n_in,
                              void* d_out, int out_size, void* d_ws, size_t ws_size,
                              hipStream_t stream) {
    const float* F        = (const float*)d_in[0];
    const int*   coords   = (const int*)d_in[1];
    const int*   nbr_idx  = (const int*)d_in[2];
    const int*   aux_idx  = (const int*)d_in[3];
    const float* counts_v = (const float*)d_in[4];
    const int*   num_aux  = (const int*)d_in[5];
    const float* alpha    = (const float*)d_in[6];
    const float* w_pos    = (const float*)d_in[7];
    const float* w_pre    = (const float*)d_in[8];
    const float* g_pre    = (const float*)d_in[9];
    const float* b_pre    = (const float*)d_in[10];
    const float* w_conv   = (const float*)d_in[11];
    const float* g_local  = (const float*)d_in[12];
    const float* b_local  = (const float*)d_in[13];
    const float* g_norm   = (const float*)d_in[14];
    const float* b_norm   = (const float*)d_in[15];

    const int N = in_sizes[0] / INC;     // 100000: divisible by 16 and 32

    // workspace layout (256B aligned)
    char* ws = (char*)d_ws;
    size_t o = 0;
    _Float16* F16 = (_Float16*)(ws + o);
    o += ((size_t)(N + 1) * INC * sizeof(_Float16) + 255) & ~(size_t)255;
    _Float16* Wpre16 = (_Float16*)(ws + o);
    o += ((size_t)INC * INC * sizeof(_Float16) + 255) & ~(size_t)255;
    _Float16* WconvT16 = (_Float16*)(ws + o);
    o += ((size_t)K27 * INC * INC * sizeof(_Float16) + 255) & ~(size_t)255;
    float* aux = (float*)(ws + o);
    o += ((size_t)N * 3 * INC * sizeof(float) + 255) & ~(size_t)255;

    float* F_input = (float*)d_out;      // reused as scratch; fully overwritten by k_final

    k_convert<<<4096, 256, 0, stream>>>(F, w_pre, w_conv, F16, Wpre16, WconvT16, N);
    k_zero_aux<<<(N * 3 * INC + 255) / 256, 256, 0, stream>>>(aux, num_aux);
    k_premix<<<N / 16, 256, 0, stream>>>(F16, Wpre16, g_pre, b_pre, F_input, N);
    k_scatter<<<(N * INC + 255) / 256, 256, 0, stream>>>(coords, aux_idx, alpha, w_pos,
                                                         F_input, aux, N);
    k_final<<<N / 32, 256, 0, stream>>>(F16, WconvT16, nbr_idx, coords, aux_idx, counts_v,
                                        alpha, w_pos, F_input, aux,
                                        g_local, b_local, g_norm, b_norm,
                                        (float*)d_out, N);
}
